// GraphAttentionLayer_45002667327855
// MI455X (gfx1250) — matrix operand order repaired and checked by reading, not compiled, not run on hardware
//
#include <hip/hip_runtime.h>
#include <hip/hip_bf16.h>

typedef __attribute__((ext_vector_type(16))) _Float16 v16h;
typedef __attribute__((ext_vector_type(8)))  float    v8f;

#define N_NODES 4096
#define N_HEADS 8
#define F_OUT   64
#define F_IN    128
#define JSTEP   32
#define LOG2E   1.4426950408889634f

// ---------------------------------------------------------------------------
// Kernel 1: fused  HW = h @ W  + tf-style reshape scramble + src/tgt dots.
// HW row r, col chunk [64b, 64b+64) maps to a single (head, n):
//   head = r>>9 ; n = ((r&511)<<3) + b ; f = c&63
// Each 64-thread block handles one (r, b) chunk: thread f computes y[f],
// LDS-reduces src/tgt (prescaled by log2(e) so the attention kernel can use
// the branchless v_exp_f32 = exp2), writes htT[head][f][n] (f16, transposed
// for WMMA B-operand loads).
// ---------------------------------------------------------------------------
__global__ __launch_bounds__(64) void gat_prep(
    const float* __restrict__ h,
    const float* __restrict__ W,
    const float* __restrict__ a,
    float* __restrict__ srcv,
    float* __restrict__ tgtv,
    _Float16* __restrict__ htT) {
  const int bid  = blockIdx.x;         // 0 .. 32767
  const int r    = bid >> 3;           // HW row 0..4095
  const int b    = bid & 7;            // 64-wide column chunk
  const int f    = threadIdx.x;        // 0..63
  const int c    = b * 64 + f;
  const int head = r >> 9;
  const int n    = ((r & 511) << 3) + b;

  const float* hr = h + (size_t)r * F_IN;
  float y = 0.0f;
#pragma unroll 8
  for (int k = 0; k < F_IN; ++k) {
    y = fmaf(hr[k], W[(size_t)k * 512 + c], y);
  }

  __shared__ float s1[64];
  __shared__ float s2[64];
  s1[f] = y * a[head * 2 * F_OUT + f];
  s2[f] = y * a[head * 2 * F_OUT + F_OUT + f];
  __syncthreads();
#pragma unroll
  for (int s = 32; s > 0; s >>= 1) {
    if (f < s) { s1[f] += s1[f + s]; s2[f] += s2[f + s]; }
    __syncthreads();
  }
  if (f == 0) {
    // prescale by log2(e): leaky_relu commutes with positive scaling, so
    // exp(leaky(src+tgt)) == exp2(leaky(src*L + tgt*L))
    srcv[head * N_NODES + n] = s1[0] * LOG2E;
    tgtv[head * N_NODES + n] = s2[0] * LOG2E;
  }
  // transposed f16 V for WMMA B-operand loads (contiguous along n)
  htT[((size_t)(head * F_OUT + f)) * N_NODES + n] = (_Float16)y;
}

// ---------------------------------------------------------------------------
// Kernel 2: fused masked-softmax attention + PV matmul (flash style, no
// running max needed since logits are O(1); masked entries contribute 0).
// Grid: 256 blocks (16 output rows each) x 256 threads (8 waves = 8 heads,
// so all heads share each adj tile through WGP$).
// Per wave: A = P tile (16x32 f16, contiguous k = 16g + s), B = V tiles from htT,
// 4x v_wmma_f32_16x16x32_f16 per j-step into 16x64 f32 accumulators.
// Inner loop is fully branchless: min/max/fma leaky, v_exp_f32 via
// __builtin_amdgcn_exp2f, v_cndmask for the adjacency mask.
// ---------------------------------------------------------------------------
__global__ __launch_bounds__(256) void gat_attn(
    const int* __restrict__ adj,
    const float* __restrict__ srcv,
    const float* __restrict__ tgtv,
    const _Float16* __restrict__ htT,
    float* __restrict__ out) {
  const int lane  = threadIdx.x & 31;
  const int head  = threadIdx.x >> 5;      // 0..7 (wave id)
  const int i0    = blockIdx.x * 16;       // row block
  const int nn    = lane & 15;             // 0..15
  const int g     = lane >> 4;             // half-wave group
  const int kbase = g * 16;                // A-layout K base for this group

  const int row   = i0 + nn;               // this lane's A-matrix row
  const float srci = srcv[head * N_NODES + row];

  const int*      adjrow = adj + (size_t)row * N_NODES;
  const float*    tgrow  = tgtv + head * N_NODES;
  const _Float16* vbase  = htT + (size_t)head * F_OUT * N_NODES;

  v8f acc0 = {}, acc1 = {}, acc2 = {}, acc3 = {};
  float lsum = 0.0f;

  for (int j0 = 0; j0 < N_NODES; j0 += JSTEP) {
    // branchless in-bounds prefetch of the adj stream a few tiles ahead
    __builtin_prefetch(adjrow + ((j0 + 4 * JSTEP) & (N_NODES - 1)) + kbase, 0, 3);

    // adjacency + tgt for this lane's 16 K slots:
    // A-layout: lane group g holds K = 0..15 + 16g
    const int4* ap  = (const int4*)(adjrow + j0 + kbase);
    const int4* ap2 = (const int4*)(adjrow + j0 + kbase + 8);
    int4 a0 = ap[0],  a1 = ap[1];
    int4 a2 = ap2[0], a3 = ap2[1];
    const float4* tp  = (const float4*)(tgrow + j0 + kbase);
    const float4* tp2 = (const float4*)(tgrow + j0 + kbase + 8);
    float4 t0 = tp[0],  t1 = tp[1];
    float4 t2 = tp2[0], t3 = tp2[1];

    const int   am[16] = {a0.x, a0.y, a0.z, a0.w, a1.x, a1.y, a1.z, a1.w,
                          a2.x, a2.y, a2.z, a2.w, a3.x, a3.y, a3.z, a3.w};
    const float tm[16] = {t0.x, t0.y, t0.z, t0.w, t1.x, t1.y, t1.z, t1.w,
                          t2.x, t2.y, t2.z, t2.w, t3.x, t3.y, t3.z, t3.w};

    // P = exp2(leaky(src'+tgt')) masked by adjacency (masked -> 0, the exact
    // limit of the -1e9 fill); pack into the 16-bit A-matrix layout.
    v16h A;
#pragma unroll
    for (int idx = 0; idx < 16; ++idx) {
      float e = srci + tm[idx];
      e = fmaf(0.2f, fminf(e, 0.0f), fmaxf(e, 0.0f));   // leaky_relu(0.2)
      float p = __builtin_amdgcn_exp2f(e);              // v_exp_f32, branchless
      p = (am[idx] > 0) ? p : 0.0f;                     // v_cndmask
      lsum += p;
      A[idx] = (_Float16)p;
    }

    // V tiles (B layout: lanes 0-15 hold K=j0..j0+15, lanes 16-31 K=+16..31;
    // column nn). htT is [head][f][n] so each lane reads 16 contiguous f16.
    const _Float16* vp = vbase + j0 + g * 16;
    v16h B0 = *(const v16h*)(vp + (size_t)(0 * 16 + nn) * N_NODES);
    v16h B1 = *(const v16h*)(vp + (size_t)(1 * 16 + nn) * N_NODES);
    v16h B2 = *(const v16h*)(vp + (size_t)(2 * 16 + nn) * N_NODES);
    v16h B3 = *(const v16h*)(vp + (size_t)(3 * 16 + nn) * N_NODES);

    acc0 = __builtin_amdgcn_wmma_f32_16x16x32_f16(false, A, false, B0,
                                                  (short)0, acc0, false, false);
    acc1 = __builtin_amdgcn_wmma_f32_16x16x32_f16(false, A, false, B1,
                                                  (short)0, acc1, false, false);
    acc2 = __builtin_amdgcn_wmma_f32_16x16x32_f16(false, A, false, B2,
                                                  (short)0, acc2, false, false);
    acc3 = __builtin_amdgcn_wmma_f32_16x16x32_f16(false, A, false, B3,
                                                  (short)0, acc3, false, false);
  }

  // combine row sums: lanes l and l^16 hold partials of row (l&15)
  lsum += __shfl_xor(lsum, 16, 32);

  // C layout: lane l, acc element r -> row (r + 8*(l>>4)), col (l&15)
#pragma unroll
  for (int r = 0; r < 8; ++r) {
    const int rowM = r + 8 * g;
    const float lr = __shfl(lsum, rowM, 32);      // row sum lives in lane rowM
    const float inv = 1.0f / fmaxf(lr, 1e-20f);   // softmax normalization
    const size_t ob = ((size_t)(head * N_NODES) + i0 + rowM) * F_OUT + nn;
    out[ob +  0] = acc0[r] * inv;
    out[ob + 16] = acc1[r] * inv;
    out[ob + 32] = acc2[r] * inv;
    out[ob + 48] = acc3[r] * inv;
  }
}

// ---------------------------------------------------------------------------
// Launch: inputs in setup_inputs() order: h, adj_mat, W, a (all fp32/int32).
// Workspace: srcv (128KB) | tgtv (128KB) | htT f16 (4MB)  ~= 4.25 MB.
// ---------------------------------------------------------------------------
extern "C" void kernel_launch(void* const* d_in, const int* in_sizes, int n_in,
                              void* d_out, int out_size, void* d_ws, size_t ws_size,
                              hipStream_t stream) {
  const float* h   = (const float*)d_in[0];
  const int*   adj = (const int*)d_in[1];
  const float* W   = (const float*)d_in[2];
  const float* a   = (const float*)d_in[3];
  float* out = (float*)d_out;

  float*    srcv = (float*)d_ws;
  float*    tgtv = srcv + N_HEADS * N_NODES;
  _Float16* htT  = (_Float16*)(tgtv + N_HEADS * N_NODES);

  // 4096 HW rows x 8 column-chunks, 64 threads each
  gat_prep<<<dim3(N_NODES * 8), dim3(64), 0, stream>>>(h, W, a, srcv, tgtv, htT);
  // 256 row-blocks x (8 heads * wave32)
  gat_attn<<<dim3(N_NODES / 16), dim3(256), 0, stream>>>(adj, srcv, tgtv, htT, out);
}
